// MixtureAttention_29463475651338
// MI455X (gfx1250) — matrix operand order repaired and checked by reading, not compiled, not run on hardware
//
#include <hip/hip_runtime.h>
#include <hip/hip_bf16.h>

// ---------------- problem constants ----------------
constexpr int kD = 1024;      // model dim
constexpr int kH = 16;        // heads
constexpr int kHD = 64;       // head dim
constexpr int kE = 8;         // experts
constexpr int kB = 2;
constexpr int kN = 2048;
constexpr int kM = 4096;
constexpr int kCHUNK = kM / kE;   // 512 keys per expert
constexpr int kBN = kB * kN;      // 4096 query rows
constexpr float kSCALE = 0.125f;  // hd^-0.5

// ---------------- CDNA5 async global->LDS path (guarded) ----------------
#if __has_builtin(__builtin_amdgcn_global_load_async_to_lds_b128) && \
    __has_builtin(__builtin_amdgcn_s_wait_asynccnt)
#define USE_ASYNC_LDS 1
#else
#define USE_ASYNC_LDS 0
#endif

typedef int v4i_ __attribute__((vector_size(16)));

__device__ static inline void copy16_g2l(const unsigned short* gsrc, unsigned short* lds) {
#if USE_ASYNC_LDS
    __builtin_amdgcn_global_load_async_to_lds_b128(
        (__attribute__((address_space(1))) v4i_*)gsrc,
        (__attribute__((address_space(3))) v4i_*)lds,
        0, 0);
#else
    *(uint4*)lds = *(const uint4*)gsrc;
#endif
}

template <int N>
__device__ static inline void wait_async() {
#if USE_ASYNC_LDS
    __builtin_amdgcn_s_wait_asynccnt(N);
#endif
}

// ---------------- WMMA types ----------------
typedef __attribute__((ext_vector_type(16))) __bf16 v16bf;
typedef __attribute__((ext_vector_type(8)))  float  v8f;

union FragBF  { v16bf v; unsigned int u[8]; unsigned short s[16]; };
union FragF32 { v8f   v; float f[8]; };

__device__ static inline unsigned short f32_to_bf16_u(float f) {
    unsigned int u = __float_as_uint(f);
    u += 0x7FFFu + ((u >> 16) & 1u);   // round-to-nearest-even
    return (unsigned short)(u >> 16);
}

__device__ static inline v8f wmma_bf16(v16bf a, v16bf b, v8f c) {
    // D = A*B + C, 16x16x32 bf16 -> f32
    return __builtin_amdgcn_wmma_f32_16x16x32_bf16(
        /*neg_a=*/false, a, /*neg_b=*/false, b,
        /*c_mod=*/(short)0, c, /*reuse_a=*/false, /*reuse_b=*/false);
}

// A fragment (16x32, 16-bit, row-major source; pairs along K contiguous).
// ISA 7.12.2: lanes 0-15 M=lane; VGPR v holds K = (v<4?0:16)+half*8+2*(v&3), +1
__device__ static inline v16bf load_frag_a(const unsigned short* A, int lda,
                                           int m0, int k0, int lane) {
    FragBF fr;
    const int half = (lane >> 4) & 1;
    const unsigned short* base = A + (long long)(m0 + (lane & 15)) * lda + k0;
#pragma unroll
    for (int v = 0; v < 8; ++v) {
        int kk = ((v & 4) ? 16 : 0) + half * 8 + ((v & 3) << 1);
        fr.u[v] = *(const unsigned int*)(base + kk);
    }
    return fr.v;
}

// B fragment (32x16, 16-bit) read from TRANSPOSED storage Bt[n][k], ldb over k.
// Layout: lane -> N = lane&15; VGPR j holds K = (j<4?0:16) + (lane>=16?8:0) + 2*(j&3), +1
__device__ static inline v16bf load_frag_b_t(const unsigned short* Bt, int ldb,
                                             int k0, int n0, int lane) {
    FragBF fr;
    const int kb = k0 + (((lane >> 4) & 1) << 3);
    const unsigned short* base = Bt + (long long)(n0 + (lane & 15)) * ldb + kb;
#pragma unroll
    for (int j = 0; j < 8; ++j) {
        fr.u[j] = *(const unsigned int*)(base + (((j & 4) ? 16 : 0) + ((j & 3) << 1)));
    }
    return fr.v;
}

// ---------------- elementwise / conversion kernels ----------------
__global__ void zero_f32_kernel(float* out, int n) {
    int i = blockIdx.x * 256 + threadIdx.x;
    if (i < n) out[i] = 0.0f;
}

__global__ void cvt_bf16_kernel(const float* __restrict__ in,
                                unsigned short* __restrict__ out, int n) {
    int i = blockIdx.x * 256 + threadIdx.x;
    if (i < n) out[i] = f32_to_bf16_u(in[i]);
}

// W[k][n] (row-major, dim x dim) -> Wt[n][k] bf16
__global__ void transpose_cvt_kernel(const float* __restrict__ W,
                                     unsigned short* __restrict__ Wt, int dim) {
    int idx = blockIdx.x * 256 + threadIdx.x;
    if (idx >= dim * dim) return;
    int k = idx / dim, n = idx % dim;
    Wt[(long long)n * dim + k] = f32_to_bf16_u(W[idx]);
}

// ---------------- router: top-2 + softmax -> dense weights [BN][E] ----------------
__global__ void router_kernel(const float* __restrict__ q,
                              const float* __restrict__ Wr,
                              const float* __restrict__ br,
                              float* __restrict__ wts) {
    const int w = threadIdx.x >> 5, lane = threadIdx.x & 31;
    const int row = blockIdx.x * 8 + w;
    if (row >= kBN) return;
    const float* qr = q + (long long)row * kD;
    float acc[kE];
#pragma unroll
    for (int e = 0; e < kE; ++e) acc[e] = 0.0f;
    for (int i = lane; i < kD; i += 32) {
        float qv = qr[i];
        const float* wr = Wr + i * kE;
#pragma unroll
        for (int e = 0; e < kE; ++e) acc[e] += qv * wr[e];
    }
#pragma unroll
    for (int off = 16; off > 0; off >>= 1) {
#pragma unroll
        for (int e = 0; e < kE; ++e) acc[e] += __shfl_down(acc[e], off, 32);
    }
    if (lane == 0) {
        float lg[kE];
#pragma unroll
        for (int e = 0; e < kE; ++e) lg[e] = acc[e] + br[e];
        int i1 = 0;
#pragma unroll
        for (int e = 1; e < kE; ++e) if (lg[e] > lg[i1]) i1 = e;
        int i2 = (i1 == 0) ? 1 : 0;
#pragma unroll
        for (int e = 0; e < kE; ++e) if (e != i1 && lg[e] > lg[i2]) i2 = e;
        float e2 = __expf(lg[i2] - lg[i1]);     // lg[i1] >= lg[i2]
        float inv = 1.0f / (1.0f + e2);
        float* wo = wts + row * kE;
#pragma unroll
        for (int e = 0; e < kE; ++e) wo[e] = 0.0f;
        wo[i1] = inv;
        wo[i2] = e2 * inv;
    }
}

// ---------------- bf16 WMMA GEMM: C[row,n] = A[row,:] . Bt[n,:] + bias[n] -------------
// MODE 0: bf16 out, natural [row][n]
// MODE 1: bf16 out, transposed per batch [n][row]   (for V -> Vt)
// MODE 2: f32  out[row][n] += wrow[row*wstride] * (acc + bias[n])   (combine)
//
// Block 256 threads = 8 waves arranged 2(M) x 4(N); wave tile 32x64.
// Block tile M=64, N=256. A k-slab (64x32 bf16, 4KB) staged in LDS via async
// copies, double-buffered; B fragments streamed from global (transposed layout).
template<int MODE>
__global__ void gemm_bf16_kernel(const unsigned short* __restrict__ A, int lda, long long a_bs,
                                 const unsigned short* __restrict__ Bt,
                                 const float* __restrict__ bias,
                                 void* __restrict__ outp, long long o_bs,
                                 int K, int Nld, int rows_pb,
                                 const float* __restrict__ wrow, int wstride) {
    (void)rows_pb; (void)wrow; (void)wstride;
    __shared__ unsigned short sA[2][64 * 32];   // two 64-row x 32-k bf16 slabs

    const int tid = threadIdx.x, w = tid >> 5, lane = tid & 31;
    const int half = (lane >> 4) & 1, l16 = lane & 15;
    const int wm = w & 1, wn = w >> 1;          // 2 M-groups x 4 N-groups
    const int batch = blockIdx.z;
    const int m0blk = blockIdx.y * 64;
    const int n0 = blockIdx.x * 256 + wn * 64;
    const unsigned short* Ab = A + (long long)batch * a_bs;

    // copy helper: one b128 per thread covers the 4KB slab
    const int crow = tid >> 2, cseg = tid & 3;
    auto stage_tile = [&](int k0s, int buf) {
        const unsigned short* src = Ab + (long long)(m0blk + crow) * lda + k0s + cseg * 8;
        copy16_g2l(src, &sA[buf][crow * 32 + cseg * 8]);
    };

    FragF32 c[2][4];
#pragma unroll
    for (int mt = 0; mt < 2; ++mt)
#pragma unroll
        for (int nt = 0; nt < 4; ++nt)
#pragma unroll
            for (int i = 0; i < 8; ++i) c[mt][nt].f[i] = 0.0f;

    stage_tile(0, 0);
    int cur = 0;
    for (int k0 = 0; k0 < K; k0 += 32) {
        const bool has_next = (k0 + 32 < K);
        if (has_next) stage_tile(k0 + 32, cur ^ 1);
        if (has_next) wait_async<1>(); else wait_async<0>();
        __syncthreads();   // sA[cur] visible to all waves

        v16bf a0 = load_frag_a(&sA[cur][0], 32, wm * 32 + 0,  0, lane);
        v16bf a1 = load_frag_a(&sA[cur][0], 32, wm * 32 + 16, 0, lane);
#pragma unroll
        for (int nt = 0; nt < 4; ++nt) {
            const unsigned short* bp = Bt + (long long)(n0 + nt * 16) * K + k0;
            if (has_next) __builtin_prefetch((const void*)(bp + 32), 0, 3);
            v16bf b = load_frag_b_t(Bt, K, k0, n0 + nt * 16, lane);
            c[0][nt].v = wmma_bf16(a0, b, c[0][nt].v);
            c[1][nt].v = wmma_bf16(a1, b, c[1][nt].v);
        }
        __syncthreads();   // everyone done reading sA[cur] before it is re-staged
        cur ^= 1;
    }

#pragma unroll
    for (int mt = 0; mt < 2; ++mt) {
#pragma unroll
        for (int nt = 0; nt < 4; ++nt) {
#pragma unroll
            for (int i = 0; i < 8; ++i) {
                int row = m0blk + wm * 32 + mt * 16 + i + 8 * half;
                int col = n0 + nt * 16 + l16;
                float val = c[mt][nt].f[i] + bias[col];
                if constexpr (MODE == 0) {
                    ((unsigned short*)outp)[(long long)batch * o_bs + (long long)row * Nld + col] =
                        f32_to_bf16_u(val);
                } else if constexpr (MODE == 1) {
                    ((unsigned short*)outp)[(long long)batch * o_bs + (long long)col * rows_pb + row] =
                        f32_to_bf16_u(val);
                } else {
                    float* o = (float*)outp;
                    o[(long long)row * Nld + col] += wrow[row * wstride] * val;
                }
            }
        }
    }
}

// ---------------- attention per expert: softmax(Q.K^T * s) @ V ----------------
// grid (B*H, N/64), block 512 (16 waves). LDS: 64x512 f32 score slab (128 KB),
// softmaxed probabilities re-packed in place as bf16 pairs.
__global__ void attn_kernel(const unsigned short* __restrict__ Qp,   // [B,N,D] bf16
                            const unsigned short* __restrict__ Kp,   // [B,CHUNK,D] bf16
                            const unsigned short* __restrict__ Vpt,  // [B,D,CHUNK] bf16 (transposed)
                            unsigned short* __restrict__ AO) {       // [B,N,D] bf16
    extern __shared__ float smem[];  // 64 * 512 floats
    const int bh = blockIdx.x;
    const int b = bh / kH, h = bh % kH;
    const int q0 = blockIdx.y * 64;
    const int tid = threadIdx.x, w = tid >> 5, lane = tid & 31;
    const int half = (lane >> 4) & 1, l16 = lane & 15;

    // ---- phase 1: S = scale * Q . K^T  (each wave: 16 rows x 128 cols) ----
    {
        const int rg = w & 3, cg = w >> 2;  // 4 row groups x 4 col groups of 128
        const unsigned short* Qbase = Qp + (long long)(b * kN + q0 + rg * 16) * kD + h * kHD;
        const unsigned short* Kbase = Kp + (long long)b * kCHUNK * kD + h * kHD;
        FragF32 s[8];
#pragma unroll
        for (int t = 0; t < 8; ++t)
#pragma unroll
            for (int i = 0; i < 8; ++i) s[t].f[i] = 0.0f;
#pragma unroll
        for (int k0 = 0; k0 < kHD; k0 += 32) {
            v16bf a = load_frag_a(Qbase, kD, 0, k0, lane);
#pragma unroll
            for (int t = 0; t < 8; ++t) {
                v16bf bf = load_frag_b_t(Kbase, kD, k0, cg * 128 + t * 16, lane);
                s[t].v = wmma_bf16(a, bf, s[t].v);
            }
        }
#pragma unroll
        for (int t = 0; t < 8; ++t)
#pragma unroll
            for (int i = 0; i < 8; ++i)
                smem[(rg * 16 + i + 8 * half) * 512 + cg * 128 + t * 16 + l16] =
                    s[t].f[i] * kSCALE;
    }
    __syncthreads();

    // ---- phase 2: row softmax, repack in place as bf16 pairs ----
    if (tid < 64) {
        float* rowp = smem + tid * 512;
        float mx = -3.4e38f;
        for (int j = 0; j < 512; ++j) mx = fmaxf(mx, rowp[j]);
        float sum = 0.0f;
        for (int j = 0; j < 512; ++j) sum += __expf(rowp[j] - mx);
        float inv = 1.0f / sum;
        unsigned int* pk = (unsigned int*)rowp;
        for (int j = 0; j < 512; j += 2) {
            unsigned short p0 = f32_to_bf16_u(__expf(rowp[j] - mx) * inv);
            unsigned short p1 = f32_to_bf16_u(__expf(rowp[j + 1] - mx) * inv);
            pk[j >> 1] = (unsigned int)p0 | ((unsigned int)p1 << 16);
        }
    }
    __syncthreads();

    // ---- phase 3: O = P @ V  (each wave: 16 rows x 16 head-dim cols) ----
    {
        const int rg = w & 3, fc = w >> 2;  // 4 row groups x 4 hd tiles
        const unsigned short* Pus = (const unsigned short*)smem;  // ldp = 1024 ushorts/row
        const unsigned short* Vbase = Vpt + (long long)b * kD * kCHUNK + (long long)(h * kHD) * kCHUNK;
        FragF32 o;
#pragma unroll
        for (int i = 0; i < 8; ++i) o.f[i] = 0.0f;
        for (int k0 = 0; k0 < kCHUNK; k0 += 32) {
            v16bf a  = load_frag_a(Pus, 1024, rg * 16, k0, lane);
            v16bf bf = load_frag_b_t(Vbase, kCHUNK, k0, fc * 16, lane);
            o.v = wmma_bf16(a, bf, o.v);
        }
#pragma unroll
        for (int i = 0; i < 8; ++i) {
            int row = q0 + rg * 16 + i + 8 * half;
            AO[(long long)(b * kN + row) * kD + h * kHD + fc * 16 + l16] = f32_to_bf16_u(o.f[i]);
        }
    }
}

// ---------------- host launcher ----------------
extern "C" void kernel_launch(void* const* d_in, const int* in_sizes, int n_in,
                              void* d_out, int out_size, void* d_ws, size_t ws_size,
                              hipStream_t stream) {
    (void)in_sizes; (void)n_in; (void)out_size; (void)ws_size;
    const float* queries = (const float*)d_in[0];
    const float* keys    = (const float*)d_in[1];
    const float* values  = (const float*)d_in[2];
    const float* Wq = (const float*)d_in[3];
    const float* bq = (const float*)d_in[4];
    const float* Wk = (const float*)d_in[5];
    const float* bk = (const float*)d_in[6];
    const float* Wv = (const float*)d_in[7];
    const float* bv = (const float*)d_in[8];
    const float* Wo = (const float*)d_in[9];
    const float* bo = (const float*)d_in[10];
    const float* Wr = (const float*)d_in[11];
    const float* br = (const float*)d_in[12];
    float* out = (float*)d_out;

    // workspace carve (~70 MB)
    char* p = (char*)d_ws;
    auto alloc = [&](size_t bytes) -> char* {
        char* r = p;
        p += (bytes + 255) & ~(size_t)255;
        return r;
    };
    unsigned short* q_bf = (unsigned short*)alloc((size_t)kBN * kD * 2);
    unsigned short* k_bf = (unsigned short*)alloc((size_t)kB * kM * kD * 2);
    unsigned short* v_bf = (unsigned short*)alloc((size_t)kB * kM * kD * 2);
    float*          wts  = (float*)alloc((size_t)kBN * kE * 4);
    unsigned short* wq_t = (unsigned short*)alloc((size_t)kD * kD * 2);
    unsigned short* wk_t = (unsigned short*)alloc((size_t)kD * kD * 2);
    unsigned short* wv_t = (unsigned short*)alloc((size_t)kD * kD * 2);
    unsigned short* wo_t = (unsigned short*)alloc((size_t)kD * kD * 2);
    unsigned short* Qp   = (unsigned short*)alloc((size_t)kBN * kD * 2);
    unsigned short* Kp   = (unsigned short*)alloc((size_t)kB * kCHUNK * kD * 2);
    unsigned short* Vpt  = (unsigned short*)alloc((size_t)kB * kCHUNK * kD * 2);
    unsigned short* AO   = (unsigned short*)alloc((size_t)kBN * kD * 2);

    (void)hipFuncSetAttribute(reinterpret_cast<const void*>(attn_kernel),
                              hipFuncAttributeMaxDynamicSharedMemorySize, 64 * 512 * 4);

    zero_f32_kernel<<<(kBN * kD) / 256, 256, 0, stream>>>(out, kBN * kD);
    cvt_bf16_kernel<<<(kBN * kD) / 256, 256, 0, stream>>>(queries, q_bf, kBN * kD);
    cvt_bf16_kernel<<<(kB * kM * kD) / 256, 256, 0, stream>>>(keys, k_bf, kB * kM * kD);
    cvt_bf16_kernel<<<(kB * kM * kD) / 256, 256, 0, stream>>>(values, v_bf, kB * kM * kD);
    router_kernel<<<kBN / 8, 256, 0, stream>>>(queries, Wr, br, wts);

    for (int e = 0; e < kE; ++e) {
        const long long wmoff = (long long)e * kD * kD;
        transpose_cvt_kernel<<<(kD * kD) / 256, 256, 0, stream>>>(Wq + wmoff, wq_t, kD);
        transpose_cvt_kernel<<<(kD * kD) / 256, 256, 0, stream>>>(Wk + wmoff, wk_t, kD);
        transpose_cvt_kernel<<<(kD * kD) / 256, 256, 0, stream>>>(Wv + wmoff, wv_t, kD);
        transpose_cvt_kernel<<<(kD * kD) / 256, 256, 0, stream>>>(Wo + wmoff, wo_t, kD);

        // Q projection: [4096,1024] x [1024,1024] -> Qp bf16
        gemm_bf16_kernel<0><<<dim3(kD / 256, kBN / 64, 1), 256, 0, stream>>>(
            q_bf, kD, 0, wq_t, bq + e * kD, Qp, 0, kD, kD, kBN, nullptr, 0);
        // K projection: per-batch chunk rows of keys -> Kp bf16 [B,CHUNK,D]
        gemm_bf16_kernel<0><<<dim3(kD / 256, kCHUNK / 64, kB), 256, 0, stream>>>(
            k_bf + (long long)e * kCHUNK * kD, kD, (long long)kM * kD,
            wk_t, bk + e * kD, Kp, (long long)kCHUNK * kD, kD, kD, kCHUNK, nullptr, 0);
        // V projection, output transposed per batch -> Vpt bf16 [B,D,CHUNK]
        gemm_bf16_kernel<1><<<dim3(kD / 256, kCHUNK / 64, kB), 256, 0, stream>>>(
            v_bf + (long long)e * kCHUNK * kD, kD, (long long)kM * kD,
            wv_t, bv + e * kD, Vpt, (long long)kD * kCHUNK, kD, kD, kCHUNK, nullptr, 0);
        // attention
        attn_kernel<<<dim3(kB * kH, kN / 64), 512, 64 * 512 * 4, stream>>>(Qp, Kp, Vpt, AO);
        // output projection + router-weighted accumulate into d_out (f32)
        gemm_bf16_kernel<2><<<dim3(kD / 256, kBN / 64, 1), 256, 0, stream>>>(
            AO, kD, 0, wo_t, bo + e * kD, out, 0, kD, kD, kBN, wts + e, kE);
    }
}
